// RPN_274877907159
// MI455X (gfx1250) — matrix operand order repaired and checked by reading, not compiled, not run on hardware
//
#include <hip/hip_runtime.h>

typedef __attribute__((ext_vector_type(16))) _Float16 v16h;
typedef __attribute__((ext_vector_type(8)))  _Float16 v8h;
typedef __attribute__((ext_vector_type(8)))  float    v8f;

#define HF 200
#define WF 200
#define SPATIAL 40000
#define CIN 256
#define COUT 256
#define KTOT 2304          // 9 taps * 256
#define NH 48              // 36 reg + 9 cls, padded to 48
#define NANCH 360000
#define TOPK 1000
#define CAND_MAX 4096
#define BBOX_CLIP 4.135166556742356f   // ln(1000/16)

union AB16 { v16h v; v8h h[2]; };

__device__ __forceinline__ unsigned keyOf(float f) {
  unsigned u = __float_as_uint(f);
  return (u & 0x80000000u) ? ~u : (u | 0x80000000u);
}

__device__ __forceinline__ v8h zero8h() {
  v8h z;
  #pragma unroll
  for (int i = 0; i < 8; ++i) z[i] = (_Float16)0.f;
  return z;
}

// im2col A fetch: one 16B segment of one tile row, halo/out-of-range -> zeros
__device__ __forceinline__ v8h loadA(const _Float16* __restrict__ featT,
                                     int m, int kk, int seg) {
  const int tap = kk >> 8, cbase = kk & 255;
  const int dy = tap / 3 - 1, dx = tap % 3 - 1;
  const int my = m / WF, mx = m % WF;
  const int ys = my + dy, xs = mx + dx;
  if (m < SPATIAL && ys >= 0 && ys < HF && xs >= 0 && xs < WF)
    return *(const v8h*)(featT + (size_t)(ys * WF + xs) * CIN + cbase + seg * 8);
  return zero8h();
}

// ---------------------------------------------------------------- pre-passes
__global__ __launch_bounds__(256)
void transpose_feat(const float* __restrict__ feat, _Float16* __restrict__ featT) {
  __shared__ float tile[32][33];
  const int s0 = blockIdx.x * 32;
  const int c0 = blockIdx.y * 32;
  for (int r = threadIdx.y; r < 32; r += 8)
    tile[r][threadIdx.x] = feat[(size_t)(c0 + r) * SPATIAL + s0 + threadIdx.x];
  __syncthreads();
  for (int r = threadIdx.y; r < 32; r += 8)
    featT[(size_t)(s0 + r) * CIN + c0 + threadIdx.x] = (_Float16)tile[threadIdx.x][r];
}

__global__ __launch_bounds__(256)
void prep_w1(const float* __restrict__ w1, _Float16* __restrict__ W1h) {
  int i = blockIdx.x * blockDim.x + threadIdx.x;
  if (i >= COUT * KTOT) return;
  int n = i / KTOT, r = i % KTOT;
  int tap = r >> 8, ci = r & 255;
  // OIHW: w1[n][ci][ky][kx] -> W1h[n][tap*256+ci]
  W1h[i] = (_Float16)w1[(size_t)n * KTOT + ci * 9 + tap];
}

__global__ __launch_bounds__(256)
void prep_wh(const float* __restrict__ rw, const float* __restrict__ rbias,
             const float* __restrict__ cw, const float* __restrict__ cbias,
             _Float16* __restrict__ Wh, float* __restrict__ rb) {
  int i = blockIdx.x * blockDim.x + threadIdx.x;
  if (i < NH * 256) {
    int n = i >> 8, c = i & 255;
    float v = 0.f;
    if (n < 36) v = rw[n * 256 + c];
    else if (n < 45) v = cw[(n - 36) * 256 + c];
    Wh[i] = (_Float16)v;
  }
  if (i < NH) {
    float v = 0.f;
    if (i < 36) v = rbias[i]; else if (i < 45) v = cbias[i - 36];
    rb[i] = v;
  }
}

// --------------------------------------------------- conv1: 3x3 implicit GEMM
// C[40000,256] = im2col(feat) x W1^T.
// Tile 128M x 256N per 512-thread (16-wave) workgroup, double-buffered LDS:
// global loads for step k+1 are issued before the 8 WMMAs of step k, staged
// registers are written to the alternate LDS buffer after compute.
__global__ __launch_bounds__(512)
void conv1_wmma(const _Float16* __restrict__ featT, const _Float16* __restrict__ W1h,
                const float* __restrict__ b1, _Float16* __restrict__ tT) {
  __shared__ _Float16 As[2][128 * 32];   // [m][k]
  __shared__ _Float16 Bs[2][256 * 32];   // [n][k]  (B^T tile)
  const int tid = threadIdx.x;
  const int lane = tid & 31, wave = tid >> 5;
  const int haf = lane >> 4, l15 = lane & 15;
  const int m0 = blockIdx.x * 128;
  const int msub = (wave & 7) * 16;
  const int nhalf = (wave >> 3) * 128;

  v8f acc[8];
  #pragma unroll
  for (int i = 0; i < 8; ++i)
    #pragma unroll
    for (int j = 0; j < 8; ++j) acc[i][j] = 0.f;

  // staging roles
  const int arow = tid >> 2;            // 0..127
  const int aseg = tid & 3;             // 16B segment
  const int brow = tid >> 1;            // 0..255 (out channel)
  const int bhalf = tid & 1;            // 32B half of a 64B row

  // ---- prologue: stage k=0 into buffer 0
  {
    v8h av = loadA(featT, m0 + arow, 0, aseg);
    const v8h* bs = (const v8h*)(W1h + (size_t)brow * KTOT + bhalf * 16);
    v8h b0 = bs[0], b1v = bs[1];
    *(v8h*)(As[0] + arow * 32 + aseg * 8) = av;
    v8h* bd = (v8h*)(Bs[0] + brow * 32 + bhalf * 16);
    bd[0] = b0; bd[1] = b1v;
  }
  __syncthreads();

  int p = 0;
  for (int kk = 0; kk < KTOT; kk += 32) {
    const bool more = (kk + 32) < KTOT;
    // ---- issue global loads for next step (latency hidden under WMMAs)
    v8h av = zero8h(), b0 = zero8h(), b1v = zero8h();
    if (more) {
      av = loadA(featT, m0 + arow, kk + 32, aseg);
      const v8h* bs = (const v8h*)(W1h + (size_t)brow * KTOT + kk + 32 + bhalf * 16);
      b0 = bs[0]; b1v = bs[1];
    }
    // ---- compute on buffer p: preload all fragments, then WMMA chain
    AB16 af;
    af.h[0] = *(const v8h*)(As[p] + (msub + l15) * 32 + 8 * haf);
    af.h[1] = *(const v8h*)(As[p] + (msub + l15) * 32 + 16 + 8 * haf);
    AB16 bf[8];
    #pragma unroll
    for (int nt = 0; nt < 8; ++nt) {
      const int nrow = nhalf + nt * 16 + l15;
      bf[nt].h[0] = *(const v8h*)(Bs[p] + nrow * 32 + 8 * haf);
      bf[nt].h[1] = *(const v8h*)(Bs[p] + nrow * 32 + 16 + 8 * haf);
    }
    #pragma unroll
    for (int nt = 0; nt < 8; ++nt)
      acc[nt] = __builtin_amdgcn_wmma_f32_16x16x32_f16(
          false, af.v, false, bf[nt].v, (short)0, acc[nt], false, false);
    // ---- write staged tile into the alternate buffer
    if (more) {
      *(v8h*)(As[p ^ 1] + arow * 32 + aseg * 8) = av;
      v8h* bd = (v8h*)(Bs[p ^ 1] + brow * 32 + bhalf * 16);
      bd[0] = b0; bd[1] = b1v;
    }
    __syncthreads();
    p ^= 1;
  }

  // epilogue: +bias, relu, store f16 NHWC
  #pragma unroll
  for (int nt = 0; nt < 8; ++nt) {
    const int n = nhalf + nt * 16 + l15;
    const float bias = b1[n];
    #pragma unroll
    for (int v = 0; v < 8; ++v) {
      const int m = m0 + msub + 8 * haf + v;
      if (m < SPATIAL) {
        float val = acc[nt][v] + bias;
        val = val > 0.f ? val : 0.f;
        tT[(size_t)m * COUT + n] = (_Float16)val;
      }
    }
  }
}

// ------------------------------------------------------ heads: 1x1 conv GEMM
// [40000,48] = tT[40000,256] x Wh^T ; n<36 -> deltas, 36..44 -> cls logits
__global__ __launch_bounds__(256)
void heads_wmma(const _Float16* __restrict__ tT, const _Float16* __restrict__ Wh,
                const float* __restrict__ rb, float* __restrict__ deltas,
                float* __restrict__ logits) {
  __shared__ _Float16 Bh[NH * 256];   // [n][k]
  const int tid = threadIdx.x;
  const int lane = tid & 31, wave = tid >> 5;
  const int haf = lane >> 4, l15 = lane & 15;
  { // stage all weights once
    const v8h* s = (const v8h*)Wh;
    v8h* d = (v8h*)Bh;
    for (int i = tid; i < NH * 256 / 8; i += 256) d[i] = s[i];
  }
  __syncthreads();

  const int m0 = blockIdx.x * 128;
  const int msub = wave * 16;
  int mrow = m0 + msub + l15;
  if (mrow >= SPATIAL) mrow = SPATIAL - 1;

  v8f acc[3];
  #pragma unroll
  for (int i = 0; i < 3; ++i)
    #pragma unroll
    for (int j = 0; j < 8; ++j) acc[i][j] = 0.f;

  #pragma unroll
  for (int kk = 0; kk < 256; kk += 32) {
    AB16 af;
    af.h[0] = *(const v8h*)(tT + (size_t)mrow * 256 + kk + 8 * haf);
    af.h[1] = *(const v8h*)(tT + (size_t)mrow * 256 + kk + 16 + 8 * haf);
    #pragma unroll
    for (int nt = 0; nt < 3; ++nt) {
      AB16 bf;
      const int nrow = nt * 16 + l15;
      bf.h[0] = *(const v8h*)(Bh + nrow * 256 + kk + 8 * haf);
      bf.h[1] = *(const v8h*)(Bh + nrow * 256 + kk + 16 + 8 * haf);
      acc[nt] = __builtin_amdgcn_wmma_f32_16x16x32_f16(
          false, af.v, false, bf.v, (short)0, acc[nt], false, false);
    }
  }
  #pragma unroll
  for (int nt = 0; nt < 3; ++nt) {
    const int n = nt * 16 + l15;
    const float bias = rb[n];
    #pragma unroll
    for (int v = 0; v < 8; ++v) {
      const int m = m0 + msub + 8 * haf + v;
      if (m < SPATIAL) {
        float val = acc[nt][v] + bias;
        if (n < 36) deltas[(size_t)m * 36 + n] = val;        // = deltas[idx*4+comp]
        else if (n < 45) logits[(size_t)m * 9 + (n - 36)] = val;
      }
    }
  }
}

// ------------------------------------------------------------- top-k machinery
__global__ void zero_u32(unsigned* p, int n) {
  int i = blockIdx.x * blockDim.x + threadIdx.x;
  if (i < n) p[i] = 0u;
}

__global__ void hist_build(const float* __restrict__ logits, unsigned* __restrict__ hist) {
  int i = blockIdx.x * blockDim.x + threadIdx.x;
  if (i >= NANCH) return;
  atomicAdd(&hist[keyOf(logits[i]) >> 16], 1u);
}

__global__ __launch_bounds__(1024)
void hist_scan(const unsigned* __restrict__ hist, unsigned* __restrict__ info) {
  __shared__ unsigned part[1024];
  const int tid = threadIdx.x;
  unsigned s = 0;
  const int base = tid * 64;
  for (int k = 0; k < 64; ++k) s += hist[base + k];
  part[tid] = s;
  __syncthreads();
  if (tid == 0) {
    unsigned cum = 0, T = 0, cAbove = 0;
    for (int t = 1023; t >= 0; --t) {
      if (cum + part[t] >= TOPK) {
        unsigned c2 = cum;
        for (int b = 63; b >= 0; --b) {
          unsigned h = hist[t * 64 + b];
          if (c2 + h >= TOPK) { T = (unsigned)(t * 64 + b); cAbove = c2; break; }
          c2 += h;
        }
        break;
      }
      cum += part[t];
    }
    info[0] = T; info[1] = cAbove;   // info[2] is the compaction counter (pre-zeroed)
  }
}

__global__ void compact_cand(const float* __restrict__ logits, unsigned* __restrict__ info,
                             unsigned long long* __restrict__ cand) {
  int i = blockIdx.x * blockDim.x + threadIdx.x;
  if (i >= NANCH) return;
  unsigned key = keyOf(logits[i]);
  if ((key >> 16) >= info[0]) {
    unsigned pos = atomicAdd(&info[2], 1u);
    if (pos < CAND_MAX)
      cand[pos] = ((unsigned long long)key << 32) | (unsigned)(~(unsigned)i);
  }
}

__global__ __launch_bounds__(1024)
void sort_decode(const unsigned* __restrict__ info, const unsigned long long* __restrict__ cand,
                 const float* __restrict__ logits, const float* __restrict__ deltas,
                 float* __restrict__ topBoxes, float* __restrict__ topScores) {
  __shared__ unsigned long long s[CAND_MAX];
  const int tid = threadIdx.x;
  unsigned n = info[2]; if (n > CAND_MAX) n = CAND_MAX;
  for (int i = tid; i < CAND_MAX; i += 1024) s[i] = (i < (int)n) ? cand[i] : 0ULL;
  __syncthreads();
  for (unsigned k = 2; k <= CAND_MAX; k <<= 1) {
    for (unsigned j = k >> 1; j > 0; j >>= 1) {
      for (unsigned i = tid; i < CAND_MAX; i += 1024) {
        unsigned ixj = i ^ j;
        if (ixj > i) {
          unsigned long long a = s[i], b = s[ixj];
          const bool desc = ((i & k) == 0);
          if (desc ? (a < b) : (a > b)) { s[i] = b; s[ixj] = a; }
        }
      }
      __syncthreads();
    }
  }
  if (tid < TOPK) {
    unsigned long long comp = s[tid];
    unsigned key = (unsigned)(comp >> 32);
    if (key == 0u) {
      topBoxes[tid * 4 + 0] = 0.f; topBoxes[tid * 4 + 1] = 0.f;
      topBoxes[tid * 4 + 2] = 0.f; topBoxes[tid * 4 + 3] = 0.f;
      topScores[tid] = 0.f;
    } else {
      const unsigned idx = ~(unsigned)comp;
      const int m = (int)(idx / 9), a = (int)(idx % 9);
      const int y = m / WF, x = m % WF;
      const int ri = a / 3, si = a % 3;
      const float scl[3] = {128.f, 256.f, 512.f};
      const float rat[3] = {0.5f, 1.f, 2.f};
      const float hr = sqrtf(rat[ri]);
      const float hs = hr * scl[si], wsc = (1.f / hr) * scl[si];
      const float sx = (float)(x * 4), sy = (float)(y * 4);
      const float ax1 = roundf(-wsc * 0.5f) + sx;
      const float ay1 = roundf(-hs * 0.5f) + sy;
      const float ax2 = roundf(wsc * 0.5f) + sx;
      const float ay2 = roundf(hs * 0.5f) + sy;
      const float wa = ax2 - ax1, ha = ay2 - ay1;
      const float cxa = ax1 + 0.5f * wa, cya = ay1 + 0.5f * ha;
      const float ddx = deltas[(size_t)idx * 4 + 0];
      const float ddy = deltas[(size_t)idx * 4 + 1];
      const float dw = fminf(deltas[(size_t)idx * 4 + 2], BBOX_CLIP);
      const float dh = fminf(deltas[(size_t)idx * 4 + 3], BBOX_CLIP);
      const float cx = ddx * wa + cxa, cy = ddy * ha + cya;
      const float bw = expf(dw) * wa, bh = expf(dh) * ha;
      topBoxes[tid * 4 + 0] = fminf(fmaxf(cx - 0.5f * bw, 0.f), 800.f);
      topBoxes[tid * 4 + 1] = fminf(fmaxf(cy - 0.5f * bh, 0.f), 800.f);
      topBoxes[tid * 4 + 2] = fminf(fmaxf(cx + 0.5f * bw, 0.f), 800.f);
      topBoxes[tid * 4 + 3] = fminf(fmaxf(cy + 0.5f * bh, 0.f), 800.f);
      topScores[tid] = 1.f / (1.f + expf(-logits[idx]));
    }
  }
}

// ----------------------------------------------------------------------- NMS
__global__ __launch_bounds__(256)
void nms_mask(const float* __restrict__ tb, unsigned* __restrict__ mask) {
  const int lane = threadIdx.x & 31;
  const int row = blockIdx.x * 8 + (threadIdx.x >> 5);
  if (row >= TOPK) return;
  const float x1i = tb[row * 4 + 0], y1i = tb[row * 4 + 1];
  const float x2i = tb[row * 4 + 2], y2i = tb[row * 4 + 3];
  const float areai = fmaxf(x2i - x1i, 0.f) * fmaxf(y2i - y1i, 0.f);
  for (int w = 0; w < 32; ++w) {
    const int j = w * 32 + lane;
    bool sup = false;
    if (j < TOPK && j > row) {
      const float x1j = tb[j * 4 + 0], y1j = tb[j * 4 + 1];
      const float x2j = tb[j * 4 + 2], y2j = tb[j * 4 + 3];
      const float areaj = fmaxf(x2j - x1j, 0.f) * fmaxf(y2j - y1j, 0.f);
      const float xx1 = fmaxf(x1i, x1j), yy1 = fmaxf(y1i, y1j);
      const float xx2 = fminf(x2i, x2j), yy2 = fminf(y2i, y2j);
      const float inter = fmaxf(xx2 - xx1, 0.f) * fmaxf(yy2 - yy1, 0.f);
      sup = (inter / (areai + areaj - inter + 1e-9f)) > 0.7f;
    }
    const unsigned word = (unsigned)__ballot(sup);
    if (lane == 0) mask[row * 32 + w] = word;
  }
}

__global__ void nms_seq(const unsigned* __restrict__ mask, unsigned* __restrict__ keep) {
  if (threadIdx.x != 0 || blockIdx.x != 0) return;
  unsigned removed[32];
  for (int w = 0; w < 32; ++w) removed[w] = 0u;
  for (int i = 0; i < TOPK; ++i) {
    if ((removed[i >> 5] >> (i & 31)) & 1u) { keep[i] = 0u; }
    else {
      keep[i] = 1u;
      for (int w = 0; w < 32; ++w) removed[w] |= mask[i * 32 + w];
    }
  }
}

__global__ void finalize(const float* __restrict__ tb, const float* __restrict__ ts,
                         const unsigned* __restrict__ keep, float* __restrict__ out) {
  int i = blockIdx.x * blockDim.x + threadIdx.x;
  if (i >= TOPK * 5) return;
  const int r = i / 5, c = i % 5;
  out[i] = keep[r] ? (c < 4 ? tb[r * 4 + c] : ts[r]) : 0.f;
}

// --------------------------------------------------------------------- launch
extern "C" void kernel_launch(void* const* d_in, const int* in_sizes, int n_in,
                              void* d_out, int out_size, void* d_ws, size_t ws_size,
                              hipStream_t stream) {
  const float* feat = (const float*)d_in[1];
  const float* w1   = (const float*)d_in[2];
  const float* b1   = (const float*)d_in[3];
  const float* regw = (const float*)d_in[4];
  const float* regb = (const float*)d_in[5];
  const float* clsw = (const float*)d_in[6];
  const float* clsb = (const float*)d_in[7];
  float* out = (float*)d_out;

  char* p = (char*)d_ws;
  auto alloc = [&](size_t bytes) { void* r = (void*)p; p += (bytes + 255) & ~(size_t)255; return r; };
  _Float16* featT = (_Float16*)alloc((size_t)SPATIAL * CIN * 2);
  _Float16* W1h   = (_Float16*)alloc((size_t)COUT * KTOT * 2);
  _Float16* tT    = (_Float16*)alloc((size_t)SPATIAL * COUT * 2);
  _Float16* Wh    = (_Float16*)alloc((size_t)NH * 256 * 2);
  float* rb       = (float*)alloc(NH * 4);
  float* deltas   = (float*)alloc((size_t)SPATIAL * 36 * 4);
  float* logits   = (float*)alloc((size_t)SPATIAL * 9 * 4);
  unsigned* hist  = (unsigned*)alloc(65536 * 4 + 64);
  unsigned* info  = hist + 65536;   // [0]=T bin, [1]=count above, [2]=cand counter
  unsigned long long* cand = (unsigned long long*)alloc((size_t)CAND_MAX * 8);
  float* topBoxes  = (float*)alloc(TOPK * 4 * 4);
  float* topScores = (float*)alloc(TOPK * 4);
  unsigned* mask   = (unsigned*)alloc(TOPK * 32 * 4);
  unsigned* keep   = (unsigned*)alloc(TOPK * 4);

  transpose_feat<<<dim3(SPATIAL / 32, CIN / 32), dim3(32, 8), 0, stream>>>(feat, featT);
  prep_w1<<<(COUT * KTOT + 255) / 256, 256, 0, stream>>>(w1, W1h);
  prep_wh<<<(NH * 256 + 255) / 256, 256, 0, stream>>>(regw, regb, clsw, clsb, Wh, rb);
  conv1_wmma<<<(SPATIAL + 127) / 128, 512, 0, stream>>>(featT, W1h, b1, tT);
  heads_wmma<<<(SPATIAL + 127) / 128, 256, 0, stream>>>(tT, Wh, rb, deltas, logits);
  zero_u32<<<(65536 + 3 + 255) / 256, 256, 0, stream>>>(hist, 65536 + 3);
  hist_build<<<(NANCH + 255) / 256, 256, 0, stream>>>(logits, hist);
  hist_scan<<<1, 1024, 0, stream>>>(hist, info);
  compact_cand<<<(NANCH + 255) / 256, 256, 0, stream>>>(logits, info, cand);
  sort_decode<<<1, 1024, 0, stream>>>(info, cand, logits, deltas, topBoxes, topScores);
  nms_mask<<<(TOPK + 7) / 8, 256, 0, stream>>>(topBoxes, mask);
  nms_seq<<<1, 64, 0, stream>>>(mask, keep);
  finalize<<<(TOPK * 5 + 255) / 256, 256, 0, stream>>>(topBoxes, topScores, keep, out);
  (void)in_sizes; (void)n_in; (void)out_size; (void)ws_size;
}
